// GaussianDecoder_27187142983803
// MI455X (gfx1250) — compile-verified, matrix operand order checked by reading, not run on hardware
//
#include <hip/hip_runtime.h>

// ============================================================================
// GaussianDecoder top-k on MI455X (gfx1250, wave32, WMMA).
// All matmuls use v_wmma_f32_16x16x32_bf16 with bf16 hi/lo operand splitting
// (3 WMMAs per K-chunk) for ~fp32 accuracy with matrix-core throughput.
// Score GEMM stages its streamed operand through LDS with
// global_load_async_to_lds_b128 (ASYNCcnt) shared by all 8 waves of a block.
// Workspace requirement: ~158 MB.
// ============================================================================

typedef __attribute__((ext_vector_type(16))) __bf16 v16bf;
typedef __attribute__((ext_vector_type(8)))  float  v8f;

#define DEV static __device__ __forceinline__

// Problem constants (fixed by setup_inputs()).
constexpr int BB = 512;    // batch
constexpr int DD = 256;    // data dim
constexpr int NN = 32768;  // mixture components
constexpr int LL = 128;    // latent dim
constexpr int HH = 512;    // hidden dim
constexpr int TD = 512;    // 2*D
constexpr int KK = 16;     // top-k
constexpr float LOG2PI = 1.8378770664093453f;
constexpr float CONST_TERM = -0.5f * (float)DD * LOG2PI;

DEV v8f zero8() {
  v8f v;
#pragma unroll
  for (int i = 0; i < 8; ++i) v[i] = 0.0f;
  return v;
}

DEV v8f wmma_bf16(v16bf a, v16bf b, v8f c) {
  // D = A(16x32 bf16) * B(32x16 bf16) + C(16x16 f32)
  return __builtin_amdgcn_wmma_f32_16x16x32_bf16(
      /*neg_a=*/false, a, /*neg_b=*/false, b,
      /*c_mod=*/(short)0, c, /*reuse_a=*/false, /*reuse_b=*/false);
}

// hi/lo split product: a*b ~= ah*bh + ah*bl + al*bh  (al*bl term negligible)
DEV v8f mma3(v16bf ah, v16bf al, v16bf bh, v16bf bl, v8f acc) {
  acc = wmma_bf16(al, bh, acc);
  acc = wmma_bf16(ah, bl, acc);
  acc = wmma_bf16(ah, bh, acc);
  return acc;
}

// Load one 16x32 bf16 fragment from a row-major [rows][ld] bf16 array.
// Per ISA 7.12.2 (16-bit A layout): lane&15 = row; lanes>=16 take K+8 groups.
// The same loader serves the B operand when the B matrix is stored with its
// N-columns as memory rows (i.e. transposed, K contiguous). Works for global
// or (after inlining) LDS-derived pointers.
DEV v16bf load_frag(const __bf16* __restrict__ base, int row0, int ld, int k0,
                    int lane) {
  const int r = row0 + (lane & 15);
  const int kb = k0 + ((lane >> 4) << 3);           // +8 for upper half-wave
  const __bf16* p = base + (size_t)r * (size_t)ld + kb;
  v16bf f;
#pragma unroll
  for (int j = 0; j < 8; ++j) {
    const int kk = ((j >> 2) << 4) + ((j & 3) << 1); // 0,2,4,6,16,18,20,22
    union { unsigned u; __bf16 h[2]; } t;
    t.u = *reinterpret_cast<const unsigned*>(p + kk); // 4B-aligned dword load
    f[2 * j]     = t.h[0];
    f[2 * j + 1] = t.h[1];
  }
  return f;
}

DEV void split_bf16(float v, __bf16* hi, __bf16* lo) {
  __bf16 h = (__bf16)v;
  *hi = h;
  *lo = (__bf16)(v - (float)h);
}

DEV float softplus_f(float v) {
  return (v > 15.0f) ? v : log1pf(expf(v));
}
DEV float std_from_logvar(float lv) {
  float s = softplus_f(lv) + 0.1f;
  return fminf(fmaxf(s, 0.1f), 1.0f);
}

// ============================================================================
// K0: build all bf16 hi/lo operand buffers.
//   zsplit[N][L]; W1t[H][L] (transposed); W2t[TD][H] (transposed);
//   Xext[B][512] = [-0.5*x^2 | x]
// ============================================================================
__global__ __launch_bounds__(256) void prep_kernel(
    const float* __restrict__ x, const float* __restrict__ z,
    const float* __restrict__ W1, const float* __restrict__ W2,
    __bf16* zhi, __bf16* zlo, __bf16* w1hi, __bf16* w1lo,
    __bf16* w2hi, __bf16* w2lo, __bf16* xhi, __bf16* xlo) {
  const int ZN = NN * LL;          // 4194304
  const int W1N = HH * LL;         // 65536
  const int W2N = TD * HH;         // 262144
  const int XN = BB * 512;         // 262144
  const int TOT = ZN + W1N + W2N + XN;
  for (int id = blockIdx.x * blockDim.x + threadIdx.x; id < TOT;
       id += gridDim.x * blockDim.x) {
    if (id < ZN) {
      split_bf16(z[id], &zhi[id], &zlo[id]);
    } else if (id < ZN + W1N) {
      int t = id - ZN;                       // t = h*L + l
      int h = t >> 7, l = t & 127;
      split_bf16(W1[l * HH + h], &w1hi[t], &w1lo[t]);
    } else if (id < ZN + W1N + W2N) {
      int t = id - ZN - W1N;                 // t = d*H + h2
      int d = t >> 9, h2 = t & 511;
      split_bf16(W2[h2 * TD + d], &w2hi[t], &w2lo[t]);
    } else {
      int t = id - ZN - W1N - W2N;           // t = b*512 + kk
      int b = t >> 9, kk = t & 511;
      float v;
      if (kk < DD) { float xv = x[b * DD + kk]; v = -0.5f * xv * xv; }
      else         { v = x[b * DD + (kk - DD)]; }
      split_bf16(v, &xhi[t], &xlo[t]);
    }
  }
}

// ============================================================================
// K1 (and K6): h = relu(in @ W1 + b1); in: [R][128], out: [R][512] hi/lo.
// One wave per 16-row tile; 32 output col-tiles; K=128 -> 4 chunks x 3 WMMA.
// ============================================================================
__global__ __launch_bounds__(256) void mlp1_kernel(
    int R, const __bf16* __restrict__ in_hi, const __bf16* __restrict__ in_lo,
    const __bf16* __restrict__ w1hi, const __bf16* __restrict__ w1lo,
    const float* __restrict__ b1, __bf16* __restrict__ out_hi,
    __bf16* __restrict__ out_lo) {
  const int wave = blockIdx.x * (blockDim.x >> 5) + (threadIdx.x >> 5);
  const int lane = threadIdx.x & 31;
  const int tiles = R >> 4;
  if (wave >= tiles) return;
  const int r0 = wave << 4;

  // A fragments reused across all 32 output tiles: hoist.
  v16bf ah[4], al[4];
#pragma unroll
  for (int c = 0; c < 4; ++c) {
    ah[c] = load_frag(in_hi, r0, LL, c * 32, lane);
    al[c] = load_frag(in_lo, r0, LL, c * 32, lane);
  }
  const int rsel = (lane >> 4) << 3;
  const int col0 = lane & 15;
  for (int j = 0; j < 32; ++j) {
    const int h0 = j << 4;
    v8f acc = zero8();
#pragma unroll
    for (int c = 0; c < 4; ++c) {
      v16bf bh = load_frag(w1hi, h0, LL, c * 32, lane);
      v16bf bl = load_frag(w1lo, h0, LL, c * 32, lane);
      acc = mma3(ah[c], al[c], bh, bl, acc);
    }
    const int col = h0 + col0;
    const float bias = b1[col];
#pragma unroll
    for (int i = 0; i < 8; ++i) {
      const int row = r0 + i + rsel;
      float v = fmaxf(acc[i] + bias, 0.0f);
      split_bf16(v, &out_hi[(size_t)row * HH + col],
                    &out_lo[(size_t)row * HH + col]);
    }
  }
}

// ============================================================================
// K2: raw = h @ W2 + b2 -> mu, logvar -> P[n][0:256]=inv_var,
//     P[n][256:512]=mu*inv_var (bf16 hi/lo), c[n] scalar via shfl reduction.
// ============================================================================
__global__ __launch_bounds__(256) void params_kernel(
    const __bf16* __restrict__ h_hi, const __bf16* __restrict__ h_lo,
    const __bf16* __restrict__ w2hi, const __bf16* __restrict__ w2lo,
    const float* __restrict__ b2, const float* __restrict__ log_w,
    __bf16* __restrict__ p_hi, __bf16* __restrict__ p_lo,
    float* __restrict__ cbuf) {
  const int wave = blockIdx.x * (blockDim.x >> 5) + (threadIdx.x >> 5);
  const int lane = threadIdx.x & 31;
  if (wave >= (NN >> 4)) return;
  const int r0 = wave << 4;
  const int rsel = (lane >> 4) << 3;
  const int col0 = lane & 15;

  float csum[8];
#pragma unroll
  for (int i = 0; i < 8; ++i) csum[i] = 0.0f;

  for (int j = 0; j < 16; ++j) {          // 16 tiles over D=256
    v8f accm = zero8(), acclv = zero8();
    for (int c = 0; c < 16; ++c) {        // K = 512
      const int k0 = c << 5;
      v16bf ah = load_frag(h_hi, r0, HH, k0, lane);
      v16bf al = load_frag(h_lo, r0, HH, k0, lane);
      v16bf bmh = load_frag(w2hi, j << 4, HH, k0, lane);
      v16bf bml = load_frag(w2lo, j << 4, HH, k0, lane);
      v16bf blh = load_frag(w2hi, DD + (j << 4), HH, k0, lane);
      v16bf bll = load_frag(w2lo, DD + (j << 4), HH, k0, lane);
      accm  = mma3(ah, al, bmh, bml, accm);
      acclv = mma3(ah, al, blh, bll, acclv);
    }
    const int d = (j << 4) + col0;
    const float bmu = b2[d], blv = b2[DD + d];
#pragma unroll
    for (int i = 0; i < 8; ++i) {
      const int n = r0 + i + rsel;
      float mu = accm[i] + bmu;
      float sd = std_from_logvar(acclv[i] + blv);
      float iv = 1.0f / (sd * sd);
      split_bf16(iv,      &p_hi[(size_t)n * 512 + d],
                          &p_lo[(size_t)n * 512 + d]);
      split_bf16(mu * iv, &p_hi[(size_t)n * 512 + DD + d],
                          &p_lo[(size_t)n * 512 + DD + d]);
      csum[i] += -0.5f * mu * mu * iv - logf(sd);
    }
  }
  // reduce over 16 lanes of each half-wave (columns of the tile)
#pragma unroll
  for (int i = 0; i < 8; ++i) {
#pragma unroll
    for (int m = 1; m <= 8; m <<= 1) csum[i] += __shfl_xor(csum[i], m, 32);
  }
  if ((lane & 15) == 0) {
#pragma unroll
    for (int i = 0; i < 8; ++i) {
      const int n = r0 + i + rsel;
      cbuf[n] = csum[i] + log_w[n] + CONST_TERM;
    }
  }
}

// ============================================================================
// K3: scores[B][N] = Xext @ P^T + c   (M=512, N=32768, K=512)
// One block per 16-column n-tile. The 32KB P tile (hi|lo, contiguous rows in
// the [N][512] layout) is staged once into LDS via async copies and shared by
// all 8 waves; each wave owns 4 b-tiles (4 accumulators) so the whole M=512
// is covered by the block and P is read from global exactly once.
// ============================================================================
__global__ __launch_bounds__(256) void scores_kernel(
    const __bf16* __restrict__ xhi, const __bf16* __restrict__ xlo,
    const __bf16* __restrict__ p_hi, const __bf16* __restrict__ p_lo,
    const float* __restrict__ cbuf, float* __restrict__ scores) {
  __shared__ __bf16 ptile[2 * 16 * 512];   // [hi 16x512 | lo 16x512] = 32 KB
  const int n0 = blockIdx.x << 4;
  const int wv = threadIdx.x >> 5;
  const int lane = threadIdx.x & 31;

  // ---- async-stage P tile (2 x 16KB contiguous) into LDS ----
  {
    const char* ghi = (const char*)(p_hi + (size_t)n0 * 512);
    const char* glo = (const char*)(p_lo + (size_t)n0 * 512);
    const unsigned lds0 = (unsigned)(size_t)(void*)ptile;
#pragma unroll
    for (int it = 0; it < 8; ++it) {            // 2048 x 16B chunks / 256 thr
      const int ci = it * 256 + threadIdx.x;
      const char* g = (ci < 1024 ? ghi + ci * 16 : glo + (ci - 1024) * 16);
      const unsigned loff = lds0 + ci * 16;
      asm volatile("global_load_async_to_lds_b128 %0, %1, off"
                   :: "v"(loff), "v"(g)
                   : "memory");
    }
    asm volatile("s_wait_asynccnt 0x0" ::: "memory");
    __syncthreads();
  }
  const __bf16* lhi = ptile;
  const __bf16* llo = ptile + 16 * 512;

  v8f acc[4];
#pragma unroll
  for (int bt = 0; bt < 4; ++bt) acc[bt] = zero8();

  for (int c = 0; c < 16; ++c) {
    const int k0 = c << 5;
    const v16bf bh = load_frag(lhi, 0, 512, k0, lane);  // ds loads
    const v16bf bl = load_frag(llo, 0, 512, k0, lane);
#pragma unroll
    for (int bt = 0; bt < 4; ++bt) {
      const int b0 = ((wv << 2) + bt) << 4;
      v16bf ah = load_frag(xhi, b0, 512, k0, lane);
      v16bf al = load_frag(xlo, b0, 512, k0, lane);
      acc[bt] = mma3(ah, al, bh, bl, acc[bt]);
    }
  }
  const int col = n0 + (lane & 15);
  const float cv = cbuf[col];
  const int rsel = (lane >> 4) << 3;
#pragma unroll
  for (int bt = 0; bt < 4; ++bt) {
    const int b0 = ((wv << 2) + bt) << 4;
#pragma unroll
    for (int i = 0; i < 8; ++i)
      scores[(size_t)(b0 + i + rsel) * NN + col] = acc[bt][i] + cv;
  }
}

// ============================================================================
// K4: per-row top-16 (descending, matching jax.lax.top_k ordering).
// ============================================================================
DEV void tk_insert(float (&val)[16], int (&ind)[16], float s, int n) {
  if (s <= val[15]) return;
  float cv = s; int ci = n;
#pragma unroll
  for (int j = 0; j < 16; ++j) {
    if (cv > val[j]) {
      float tv = val[j]; int ti = ind[j];
      val[j] = cv; ind[j] = ci; cv = tv; ci = ti;
    }
  }
}

__global__ __launch_bounds__(256) void topk_kernel(
    const float* __restrict__ scores, int* __restrict__ idxout) {
  __shared__ float sv[256 * 16];
  __shared__ int   si[256 * 16];
  __shared__ float sv2[256];
  __shared__ int   si2[256];
  const int b = blockIdx.x, tid = threadIdx.x;
  float val[16]; int ind[16];
#pragma unroll
  for (int j = 0; j < 16; ++j) { val[j] = -3.402823466e38f; ind[j] = 0; }
  const float* row = scores + (size_t)b * NN;
  for (int n = tid; n < NN; n += 256) tk_insert(val, ind, row[n], n);
#pragma unroll
  for (int j = 0; j < 16; ++j) { sv[tid * 16 + j] = val[j]; si[tid * 16 + j] = ind[j]; }
  __syncthreads();
  if (tid < 16) {
#pragma unroll
    for (int j = 0; j < 16; ++j) { val[j] = -3.402823466e38f; ind[j] = 0; }
    for (int e = 0; e < 256; ++e) {
      const int s = tid * 256 + e;
      tk_insert(val, ind, sv[s], si[s]);
    }
#pragma unroll
    for (int j = 0; j < 16; ++j) { sv2[tid * 16 + j] = val[j]; si2[tid * 16 + j] = ind[j]; }
  }
  __syncthreads();
  if (tid == 0) {
#pragma unroll
    for (int j = 0; j < 16; ++j) { val[j] = -3.402823466e38f; ind[j] = 0; }
    for (int e = 0; e < 256; ++e) tk_insert(val, ind, sv2[e], si2[e]);
#pragma unroll
    for (int j = 0; j < 16; ++j) idxout[b * KK + j] = ind[j];
  }
}

// ============================================================================
// K5: gather z_top rows (already-split bf16) by top-k indices.
// ============================================================================
__global__ __launch_bounds__(256) void gather_kernel(
    const int* __restrict__ idx, const __bf16* __restrict__ zhi,
    const __bf16* __restrict__ zlo, __bf16* __restrict__ zthi,
    __bf16* __restrict__ ztlo) {
  const int TOT = BB * KK * LL;            // 1048576
  for (int id = blockIdx.x * blockDim.x + threadIdx.x; id < TOT;
       id += gridDim.x * blockDim.x) {
    const int r = id >> 7, l = id & 127;
    const int src = idx[r];
    zthi[id] = zhi[(size_t)src * LL + l];
    ztlo[id] = zlo[(size_t)src * LL + l];
  }
}

// ============================================================================
// K7: stage-2: raw = h_top @ W2 + b2; NLL against x (one b per 16-row tile,
// rows are the 16 top-k entries of that b). out[b][k].
// ============================================================================
__global__ __launch_bounds__(256) void stage2_kernel(
    const __bf16* __restrict__ h_hi, const __bf16* __restrict__ h_lo,
    const __bf16* __restrict__ w2hi, const __bf16* __restrict__ w2lo,
    const float* __restrict__ b2, const float* __restrict__ x,
    float* __restrict__ out) {
  const int wave = blockIdx.x * (blockDim.x >> 5) + (threadIdx.x >> 5);
  const int lane = threadIdx.x & 31;
  if (wave >= BB) return;
  const int b = wave;
  const int r0 = b << 4;                   // 16 rows of h_top for this b
  const int rsel = (lane >> 4) << 3;
  const int col0 = lane & 15;

  float csum[8];
#pragma unroll
  for (int i = 0; i < 8; ++i) csum[i] = 0.0f;

  for (int j = 0; j < 16; ++j) {
    v8f accm = zero8(), acclv = zero8();
    for (int c = 0; c < 16; ++c) {
      const int k0 = c << 5;
      v16bf ah = load_frag(h_hi, r0, HH, k0, lane);
      v16bf al = load_frag(h_lo, r0, HH, k0, lane);
      v16bf bmh = load_frag(w2hi, j << 4, HH, k0, lane);
      v16bf bml = load_frag(w2lo, j << 4, HH, k0, lane);
      v16bf blh = load_frag(w2hi, DD + (j << 4), HH, k0, lane);
      v16bf bll = load_frag(w2lo, DD + (j << 4), HH, k0, lane);
      accm  = mma3(ah, al, bmh, bml, accm);
      acclv = mma3(ah, al, blh, bll, acclv);
    }
    const int d = (j << 4) + col0;
    const float bmu = b2[d], blv = b2[DD + d];
    const float xf = x[b * DD + d];
#pragma unroll
    for (int i = 0; i < 8; ++i) {
      float mu = accm[i] + bmu;
      float sd = std_from_logvar(acclv[i] + blv);
      float df = (xf - mu) / sd;
      csum[i] += -0.5f * df * df - logf(sd);
    }
  }
#pragma unroll
  for (int i = 0; i < 8; ++i) {
#pragma unroll
    for (int m = 1; m <= 8; m <<= 1) csum[i] += __shfl_xor(csum[i], m, 32);
  }
  if ((lane & 15) == 0) {
#pragma unroll
    for (int i = 0; i < 8; ++i)
      out[b * KK + i + rsel] = csum[i] + CONST_TERM;
  }
}

// ============================================================================
// Launch
// ============================================================================
extern "C" void kernel_launch(void* const* d_in, const int* /*in_sizes*/,
                              int /*n_in*/, void* d_out, int /*out_size*/,
                              void* d_ws, size_t /*ws_size*/,
                              hipStream_t stream) {
  const float* x     = (const float*)d_in[0];
  const float* log_w = (const float*)d_in[1];
  const float* z     = (const float*)d_in[2];
  const float* W1    = (const float*)d_in[3];
  const float* b1    = (const float*)d_in[4];
  const float* W2    = (const float*)d_in[5];
  const float* b2    = (const float*)d_in[6];
  float* out = (float*)d_out;

  char* base = (char*)d_ws;
  size_t o = 0;
  auto take = [&](size_t bytes) {
    char* p = base + o;
    o += (bytes + 255) & ~(size_t)255;
    return p;
  };
  __bf16* zhi  = (__bf16*)take((size_t)NN * LL * 2);
  __bf16* zlo  = (__bf16*)take((size_t)NN * LL * 2);
  __bf16* w1hi = (__bf16*)take((size_t)HH * LL * 2);
  __bf16* w1lo = (__bf16*)take((size_t)HH * LL * 2);
  __bf16* w2hi = (__bf16*)take((size_t)TD * HH * 2);
  __bf16* w2lo = (__bf16*)take((size_t)TD * HH * 2);
  __bf16* xhi  = (__bf16*)take((size_t)BB * 512 * 2);
  __bf16* xlo  = (__bf16*)take((size_t)BB * 512 * 2);
  __bf16* phi  = (__bf16*)take((size_t)NN * 512 * 2);
  __bf16* plo  = (__bf16*)take((size_t)NN * 512 * 2);
  float*  cbuf = (float*)take((size_t)NN * 4);
  int*    idx  = (int*)take((size_t)BB * KK * 4);
  __bf16* zthi = (__bf16*)take((size_t)BB * KK * LL * 2);
  __bf16* ztlo = (__bf16*)take((size_t)BB * KK * LL * 2);
  // Phase-overlapped region (64 MiB): stage-1 h (hi|lo) -> scores -> h_top.
  char* regionA = take((size_t)NN * HH * 2 * 2);
  __bf16* h1hi = (__bf16*)regionA;
  __bf16* h1lo = (__bf16*)(regionA + (size_t)NN * HH * 2);
  float*  scor = (float*)regionA;
  __bf16* hthi = (__bf16*)regionA;
  __bf16* htlo = (__bf16*)(regionA + (size_t)BB * KK * HH * 2);

  // K0: operand prep/splitting
  prep_kernel<<<18688, 256, 0, stream>>>(x, z, W1, W2, zhi, zlo, w1hi, w1lo,
                                         w2hi, w2lo, xhi, xlo);
  // K1: stage-1 hidden layer (2048 tiles, 8 waves/block)
  mlp1_kernel<<<256, 256, 0, stream>>>(NN, zhi, zlo, w1hi, w1lo, b1, h1hi, h1lo);
  // K2: per-component Gaussian params -> P, c
  params_kernel<<<256, 256, 0, stream>>>(h1hi, h1lo, w2hi, w2lo, b2, log_w,
                                         phi, plo, cbuf);
  // K3: score GEMM; one block per n-tile, P staged via async LDS copy
  scores_kernel<<<NN / 16, 256, 0, stream>>>(xhi, xlo, phi, plo, cbuf, scor);
  // K4: per-row top-16
  topk_kernel<<<BB, 256, 0, stream>>>(scor, idx);
  // K5: gather z_top (split form)
  gather_kernel<<<4096, 256, 0, stream>>>(idx, zhi, zlo, zthi, ztlo);
  // K6: stage-2 hidden layer (scores are dead; h_top overwrites regionA)
  mlp1_kernel<<<64, 256, 0, stream>>>(BB * KK, zthi, ztlo, w1hi, w1lo, b1,
                                      hthi, htlo);
  // K7: stage-2 NLL
  stage2_kernel<<<64, 256, 0, stream>>>(hthi, htlo, w2hi, w2lo, b2, x, out);
}
